// FastFoodRandomFeatures_7078106103911
// MI455X (gfx1250) — matrix operand order repaired and checked by reading, not compiled, not run on hardware
//
#include <hip/hip_runtime.h>

// CDNA5 WMMA operand types
typedef __attribute__((ext_vector_type(16))) _Float16 v16h;
typedef __attribute__((ext_vector_type(8)))  _Float16 v8h;
typedef __attribute__((ext_vector_type(8)))  float    v8f;

// Tensor DMA descriptor operand types (SGPR groups)
typedef __attribute__((ext_vector_type(4))) unsigned int u32x4;
typedef __attribute__((ext_vector_type(8))) int          i32x8;
typedef __attribute__((ext_vector_type(4))) int          i32x4;

#if defined(__has_builtin)
#if __has_builtin(__builtin_amdgcn_tensor_load_to_lds) && \
    __has_builtin(__builtin_amdgcn_tensor_store_from_lds)
#define HAVE_TDM 1
#endif
#endif
#ifndef HAVE_TDM
#define HAVE_TDM 0
#endif

// Compile-time scheduling fence: hardware keeps same-wave DS ops in order;
// this stops the compiler reordering LDS loads/stores across staging phases.
#define LDS_PHASE_FENCE() asm volatile("" ::: "memory")

// ---------------------------------------------------------------------------
// Prep: build the two folded 64x64 matrices in f16, stored K-contiguous
// (transposed) so each WMMA B-fragment is two contiguous 16-byte runs.
//   m1t[n*64+k] = sqrt(1/8) * B[k] * H[k][n]
//   m2t[j*64+m] = (1/8) * S[j] * G[Pinv[m]] * H[Pinv[m]][j]
// with H[a][b] = (-1)^popcount(a & b)  (natural-order Walsh-Hadamard).
// ---------------------------------------------------------------------------
__global__ __launch_bounds__(64) void ff_prep(const float* __restrict__ Bv,
                                              const float* __restrict__ Gv,
                                              const float* __restrict__ Sv,
                                              const int*   __restrict__ Pv,
                                              _Float16* __restrict__ mats) {
  __shared__ int pinv[64];
  const int t = threadIdx.x;                  // t = output column (n or j)
  pinv[Pv[t]] = t;
  __syncthreads();

  const float c1 = 0.35355339059327373f;      // sqrt(softmax_temp) = sqrt(1/8)
  const float cf = 0.125f;                    // sqrt(1/D) = 1/8
  _Float16* __restrict__ m1t = mats;          // 64*64 halves
  _Float16* __restrict__ m2t = mats + 64 * 64;
  const float sj = Sv[t];

  for (int k = 0; k < 64; ++k) {
    const float h1 = (__builtin_popcount(k & t) & 1) ? -1.0f : 1.0f;
    m1t[t * 64 + k] = (_Float16)(c1 * Bv[k] * h1);
    const int pi = pinv[k];                   // k plays the role of m here
    const float h2 = (__builtin_popcount(pi & t) & 1) ? -1.0f : 1.0f;
    m2t[t * 64 + k] = (_Float16)(cf * sj * Gv[pi] * h2);
  }
}

// Build a WMMA f16 B-fragment (32x16 K-major tile) from a K-contiguous
// f16 matrix in LDS. Per the CDNA5 layout, VGPR r of lane l holds
// K = {0..7, 16..23} (+8 for lanes 16-31) of column N = l&15.
__device__ __forceinline__ v16h load_bfrag(const _Float16* mt, int col, int koff) {
  const v8h* p = (const v8h*)(mt + col * 64 + koff);  // 16B-aligned
  const v8h lo = p[0];   // K = koff .. koff+7
  const v8h hi = p[2];   // K = koff+16 .. koff+23
  v16h r;
#pragma unroll
  for (int i = 0; i < 8; ++i) { r[i] = lo[i]; r[8 + i] = hi[i]; }
  return r;
}

#if HAVE_TDM
// Build TDM descriptor group0: count=1, lds byte offset, 57-bit global addr,
// type=2 ("image") in bits [127:126].
__device__ __forceinline__ u32x4 tdm_group0(unsigned long long ga, unsigned int lds) {
  u32x4 g0;
  g0[0] = 1u;                                             // count=1, user mode
  g0[1] = lds;                                            // lds_addr (bytes)
  g0[2] = (unsigned int)ga;                               // global_addr[31:0]
  g0[3] = ((unsigned int)(ga >> 32) & 0x01FFFFFFu)        // global_addr[56:32]
          | 0x80000000u;                                  // type=2
  return g0;
}

// Group1 for a 2D f32 tile: 64-wide rows, 128 rows per tile, row stride 64,
// tensor_dim1 = in-bounds rows (hardware zero-fills OOB loads, drops OOB stores).
__device__ __forceinline__ i32x8 tdm_group1(int validRows) {
  i32x8 g1 = {0, 0, 0, 0, 0, 0, 0, 0};
  g1[0] = (int)(2u << 16);                 // data_size=2 (4 bytes); mask/flags=0
  g1[1] = (int)(64u << 16);                // tensor_dim0[15:0]=64 in [31:16]
  g1[2] = (int)(((unsigned)validRows & 0xFFFFu) << 16);  // tensor_dim1[15:0]
  g1[3] = (int)(64u << 16);                // tensor_dim1[31:16]=0 | tile_dim0=64
  g1[4] = 128;                             // tile_dim1=128, tile_dim2=0
  g1[5] = 64;                              // tensor_dim0_stride=64
  g1[6] = 0;                               // stride hi / dim1_stride lo
  g1[7] = 0;
  return g1;
}

__device__ __forceinline__ void tdm_load(u32x4 g0, i32x8 g1) {
  const i32x4 z4 = {0, 0, 0, 0};
#if defined(__clang_major__) && __clang_major__ >= 23
  const i32x8 z8 = {0, 0, 0, 0, 0, 0, 0, 0};
  __builtin_amdgcn_tensor_load_to_lds(g0, g1, z4, z4, z8, 0);
#else
  __builtin_amdgcn_tensor_load_to_lds(g0, g1, z4, z4, 0);
#endif
}

__device__ __forceinline__ void tdm_store(u32x4 g0, i32x8 g1) {
  const i32x4 z4 = {0, 0, 0, 0};
#if defined(__clang_major__) && __clang_major__ >= 23
  const i32x8 z8 = {0, 0, 0, 0, 0, 0, 0, 0};
  __builtin_amdgcn_tensor_store_from_lds(g0, g1, z4, z4, z8, 0);
#else
  __builtin_amdgcn_tensor_store_from_lds(g0, g1, z4, z4, 0);
#endif
}
#endif  // HAVE_TDM

// ---------------------------------------------------------------------------
// Main: 128 rows per block (8 waves x 16-row tiles).
//   TDM DMAs the 32 KiB f32 block tile global->LDS, each wave runs
//   out_tile = X_tile @ M1 @ M2 with v_wmma_f32_16x16x32_f16, results are
//   staged back to LDS and TDM DMAs them LDS->global.
// ---------------------------------------------------------------------------
__global__ __launch_bounds__(256) void ff_main(const float* __restrict__ x,
                                               float* __restrict__ out,
                                               const _Float16* __restrict__ mats,
                                               int rows) {
  __shared__ _Float16 m1t[64 * 64];          // 8 KiB
  __shared__ _Float16 m2t[64 * 64];          // 8 KiB
  __shared__ float    stage[128 * 64];       // 32 KiB block tile (in, then out)
  __shared__ _Float16 sth[8][16 * 64];       // 2 KiB per wave (W1 f16 tile)

  const int tid = threadIdx.x;
  const long rowBase = (long)blockIdx.x * 128;
  int rem = rows - (int)rowBase;             // uniform; >0 by construction
  if (rem > 128) rem = 128;

  // Cooperative copy of both matrices (16 KiB) into LDS, b128 granularity.
  {
    const uint4* __restrict__ src = (const uint4*)mats;   // 1024 uint4 total
    uint4* d1 = (uint4*)m1t;
    uint4* d2 = (uint4*)m2t;
    d1[tid]       = src[tid];
    d1[tid + 256] = src[tid + 256];
    d2[tid]       = src[tid + 512];
    d2[tid + 256] = src[tid + 768];
  }

  // ---- Phase 1: block input tile global -> LDS ----
#if HAVE_TDM
  if (tid < 32) {  // wave 0 issues the DMA (EXEC-independent, one per wave)
    const unsigned long long ga =
        (unsigned long long)(uintptr_t)(x + rowBase * 64);
    const unsigned int lds = (unsigned int)(uintptr_t)&stage[0];
    tdm_load(tdm_group0(ga, lds), tdm_group1(rem));
    __builtin_amdgcn_s_wait_tensorcnt(0);
  }
#else
  {
    const uint4* __restrict__ gx = (const uint4*)(x + rowBase * 64);
    uint4* sv = (uint4*)stage;
#pragma unroll
    for (int i = 0; i < 8; ++i) sv[i * 256 + tid] = gx[i * 256 + tid];
  }
#endif
  __syncthreads();   // publishes matrices + input tile to all waves

  const int wave  = tid >> 5;
  const int lane  = tid & 31;
  const int laneN = lane & 15;                 // A-row / C-column within tile
  const int khalf = (lane < 16) ? 0 : 8;       // A-fragment K sub-base
  const int koffB = (lane < 16) ? 0 : 8;       // B-fragment K base
  const int crow  = (lane < 16) ? 0 : 8;       // C/D row base

  float*    st = stage + wave * (16 * 64);     // this wave's 16x64 f32 slice
  _Float16* sh = sth[wave];

  // ---- Phase 2: gather A fragments (f32 -> f16) per the 16x32 A layout. ----
  // lanes 0-15: row laneN, K in {0..7, 16..23} (+32*c); lanes 16-31: +8.
  v16h a1[2];
#pragma unroll
  for (int c = 0; c < 2; ++c) {
    const float* rb = st + laneN * 64 + c * 32 + khalf;
#pragma unroll
    for (int i = 0; i < 8; ++i) {
      a1[c][i]     = (_Float16)rb[i];
      a1[c][8 + i] = (_Float16)rb[16 + i];
    }
  }

  // ---- GEMM1: W1(16x64) = X @ M1 ; 4 N-tiles x 2 K-chunks = 8 WMMAs. ----
  v8f w[4];
#pragma unroll
  for (int nt = 0; nt < 4; ++nt) {
    v8f acc = {};
    acc = __builtin_amdgcn_wmma_f32_16x16x32_f16(
        false, a1[0], false, load_bfrag(m1t, nt * 16 + laneN, koffB),
        (short)0, acc, false, false);
    acc = __builtin_amdgcn_wmma_f32_16x16x32_f16(
        false, a1[1], false, load_bfrag(m1t, nt * 16 + laneN, koffB + 32),
        (short)0, acc, false, false);
    w[nt] = acc;
  }

  // ---- Phase 3: restage W1 as f16 through LDS (C layout -> row major). ----
#pragma unroll
  for (int nt = 0; nt < 4; ++nt)
#pragma unroll
    for (int r = 0; r < 8; ++r)
      sh[(crow + r) * 64 + nt * 16 + laneN] = (_Float16)w[nt][r];
  LDS_PHASE_FENCE();

  // ---- Phase 4: gather A fragments for GEMM2 (already f16). ----
  v16h a2[2];
#pragma unroll
  for (int c = 0; c < 2; ++c) {
    const v8h* rb = (const v8h*)(sh + laneN * 64 + c * 32 + khalf);
    const v8h lo = rb[0];   // K = base .. base+7
    const v8h hi = rb[2];   // K = base+16 .. base+23
#pragma unroll
    for (int i = 0; i < 8; ++i) { a2[c][i] = lo[i]; a2[c][8 + i] = hi[i]; }
  }

  // ---- GEMM2: OUT(16x64) = W1 @ M2 ; another 8 WMMAs. ----
  v8f o[4];
#pragma unroll
  for (int nt = 0; nt < 4; ++nt) {
    v8f acc = {};
    acc = __builtin_amdgcn_wmma_f32_16x16x32_f16(
        false, a2[0], false, load_bfrag(m2t, nt * 16 + laneN, koffB),
        (short)0, acc, false, false);
    acc = __builtin_amdgcn_wmma_f32_16x16x32_f16(
        false, a2[1], false, load_bfrag(m2t, nt * 16 + laneN, koffB + 32),
        (short)0, acc, false, false);
    o[nt] = acc;
  }

  // ---- Phase 5: restage f32 result over the input slice (own slice only,
  //      same-wave DS ops are in order; input was fully consumed above). ----
  LDS_PHASE_FENCE();
#pragma unroll
  for (int nt = 0; nt < 4; ++nt)
#pragma unroll
    for (int r = 0; r < 8; ++r)
      st[(crow + r) * 64 + nt * 16 + laneN] = o[nt][r];

  __syncthreads();   // all waves' results resident in LDS

  // ---- Phase 6: block output tile LDS -> global ----
#if HAVE_TDM
  if (tid < 32) {
    const unsigned long long ga =
        (unsigned long long)(uintptr_t)(out + rowBase * 64);
    const unsigned int lds = (unsigned int)(uintptr_t)&stage[0];
    tdm_store(tdm_group0(ga, lds), tdm_group1(rem));
    __builtin_amdgcn_s_wait_tensorcnt(0);
  }
#else
  {
    const uint4* sv = (const uint4*)stage;
    uint4* __restrict__ go = (uint4*)(out + rowBase * 64);
#pragma unroll
    for (int i = 0; i < 8; ++i) go[i * 256 + tid] = sv[i * 256 + tid];
  }
#endif
}

// ---------------------------------------------------------------------------
extern "C" void kernel_launch(void* const* d_in, const int* in_sizes, int n_in,
                              void* d_out, int out_size, void* d_ws, size_t ws_size,
                              hipStream_t stream) {
  (void)n_in; (void)out_size; (void)ws_size;
  const float* x = (const float*)d_in[0];   // (N,L,H,64) f32
  const float* B = (const float*)d_in[1];   // (64,) f32
  const float* G = (const float*)d_in[2];   // (64,) f32
  const float* S = (const float*)d_in[3];   // (64,) f32
  const int*   P = (const int*)d_in[4];     // (64,) i32
  float* out = (float*)d_out;
  _Float16* mats = (_Float16*)d_ws;         // 16 KiB: M1^T then M2^T, f16

  const int rows = in_sizes[0] / 64;        // 524288
  ff_prep<<<1, 64, 0, stream>>>(B, G, S, P, mats);
  const int blocks = (rows + 127) / 128;    // 128 rows (8 wave-tiles) per block
  ff_main<<<blocks, 256, 0, stream>>>(x, out, mats, rows);
}
